// HyperNetFusion_56805237457034
// MI455X (gfx1250) — matrix operand order repaired and checked by reading, not compiled, not run on hardware
//
#include <hip/hip_runtime.h>

#define DEV __device__ __forceinline__

typedef float  v8f   __attribute__((ext_vector_type(8)));
typedef float  v16f  __attribute__((ext_vector_type(16)));
typedef __bf16 v16bf __attribute__((ext_vector_type(16)));

constexpr int Dd  = 256;
constexpr int HHk = 512;
constexpr int Hh  = 64;
constexpr int Cc  = 16;
constexpr int Tt  = 10;
constexpr int I1  = Dd * Hh;       // 16384
constexpr int I2  = I1 + Hh;       // 16448
constexpr int I3  = I2 + Hh * Cc;  // 17472
constexpr int Pp  = I3 + Cc;       // 17488
constexpr int Bb  = 512;

DEV v8f wmma_bf16(v16bf a, v16bf b, v8f c) {
  return __builtin_amdgcn_wmma_f32_16x16x32_bf16(false, a, false, b, (short)0, c,
                                                 false, false);
}

// A fragment (this code's contiguous K order, same as B): per-lane elements 0..15 = K sel+0..15
// (caller bakes sel = hi?16:0 into p).
DEV v16bf cvt_fragA(const float* p) {
  const float4* q = (const float4*)p;
  float4 a = q[0], b = q[1], c = q[2], d = q[3];
  v16f f = {a.x, a.y, a.z, a.w, b.x, b.y, b.z, b.w,
            c.x, c.y, c.z, c.w, d.x, d.y, d.z, d.w};
  return __builtin_convertvector(f, v16bf);
}

// B fragment from f32 global: 16 consecutive K values (lanes 0-15: K0..15,
// lanes 16-31: K16..31; caller bakes the +16 into p).
DEV v16bf cvt_fragB(const float* p) {
  const float4* q = (const float4*)p;
  float4 a = q[0], b = q[1], c = q[2], d = q[3];
  v16f f = {a.x, a.y, a.z, a.w, b.x, b.y, b.z, b.w,
            c.x, c.y, c.z, c.w, d.x, d.y, d.z, d.w};
  return __builtin_convertvector(f, v16bf);
}

// B fragment from LDS bf16: 16 consecutive bf16 (two ds_load_b128).
DEV v16bf ldsB(const __bf16* p) {
  union { v16bf v; uint4 q[2]; } u;
  u.q[0] = *(const uint4*)(p);
  u.q[1] = *(const uint4*)(p + 8);
  return u.v;
}

union F8 { float4 q[2]; float f[8]; };

__global__ __launch_bounds__(256) void zero_f32(float* p, int n) {
  int i = blockIdx.x * blockDim.x + threadIdx.x;
  if (i < n) p[i] = 0.0f;
}

// out[b][o] = relu(act[b,:] . wgt[o,:] + bias[o]); O fixed at 512, K in {256,512}.
// One 16x16 output tile per wave; f32 inputs converted to bf16 for WMMA.
// Output type selected at compile time (no per-element branching).
template <bool BF16OUT>
__global__ __launch_bounds__(256) void gemm_bias_relu(
    const float* __restrict__ act, const float* __restrict__ wgt,
    const float* __restrict__ bias, int K, int O,
    float* __restrict__ outF, __bf16* __restrict__ outB) {
  const int tid = threadIdx.x;
  const int w = tid >> 5, lane = tid & 31;
  const int lr = lane & 15, hi = lane >> 4;
  const int sel = hi ? 16 : 0;
  const int id = blockIdx.x * 8 + w;   // 1024 tiles total
  const int om = id & 31;              // o-tile (O/16 == 32)
  const int bn = id >> 5;              // b-tile
  const long ar = (long)(om * 16 + lr) * K + sel;
  const long br = (long)(bn * 16 + lr) * K + hi * 16;
  v8f acc = {};
  for (int s = 0; s < (K >> 5); ++s) {
    v16bf af = cvt_fragA(wgt + ar + s * 32);
    v16bf bf = cvt_fragB(act + br + s * 32);
    acc = wmma_bf16(af, bf, acc);
  }
  const int bbi = bn * 16 + lr;
#pragma unroll
  for (int j = 0; j < 8; ++j) {
    const int o = om * 16 + (hi << 3) + j;
    float v = fmaxf(acc[j] + bias[o], 0.0f);
    if constexpr (BF16OUT) outB[(long)bbi * O + o] = (__bf16)v;
    else                   outF[(long)bbi * O + o] = v;
  }
}

// Fused hypernet: one workgroup per (t, 32-sample tile). Never materializes
// all_params: each 16x16 f32 WMMA tile of (w3 . h2^T) is consumed in registers.
__global__ __launch_bounds__(256) void hyper_main(
    const float* __restrict__ x, const float* __restrict__ w3,
    const float* __restrict__ b3, const __bf16* __restrict__ h2g,
    float* __restrict__ out) {
  __shared__ float  x_s[32][260];     // padded: kills n-lane bank conflicts
  __shared__ __bf16 h2_s[32][520];    // padded, rows stay 16B aligned
  __shared__ float  th_s[64][34];
  __shared__ float  s4_s[16][34];

  const int t    = blockIdx.x >> 4;   // 10 t's x 16 b-groups
  const int b0   = (blockIdx.x & 15) * 32;
  const int tid  = threadIdx.x;
  const int w    = tid >> 5, lane = tid & 31;
  const int lr   = lane & 15, hi = lane >> 4;
  const int sel  = hi ? 16 : 0;
  const long tP  = (long)t * Pp;

  // Stage x (f32) and h2 (bf16) tiles for 32 samples.
  for (int i = tid; i < 32 * 64; i += 256) {
    int r = i >> 6, c4 = (i & 63) << 2;
    *(float4*)&x_s[r][c4] = *(const float4*)&x[(long)(b0 + r) * Dd + c4];
  }
  for (int i = tid; i < 32 * 64; i += 256) {
    int r = i >> 6, c8 = (i & 63) << 3;
    *(uint4*)&h2_s[r][c8] = *(const uint4*)&h2g[(long)(b0 + r) * HHk + c8];
  }
  __syncthreads();

  // ---- fc1 bias tile (S2): rows tP+I1+(w>>1)*16 .. +15, kept in registers.
  v8f s2a = {}, s2b = {};
  {
    const long rb = (tP + I1 + ((long)(w >> 1) << 4) + lr) * HHk + sel;
#pragma unroll
    for (int s = 0; s < 16; ++s) {
      v16bf af = cvt_fragA(w3 + rb + s * 32);
      s2a = wmma_bf16(af, ldsB(&h2_s[lr][s * 32 + hi * 16]), s2a);
      s2b = wmma_bf16(af, ldsB(&h2_s[16 + lr][s * 32 + hi * 16]), s2b);
    }
  }

  // ---- Phase A: th[h][bb] = relu(sum_d (fc1_w)*x + fc1_b). Wave w owns h=8w..8w+7.
  for (int hl = 0; hl < 8; ++hl) {
    const int h = (w << 3) + hl;
    float th0 = 0.0f, th1 = 0.0f;
    for (int tt = 0; tt < 16; ++tt) {
      const long r0 = tP + (long)h * Dd + tt * 16;
      const float* ap = w3 + (r0 + lr) * HHk + sel;
      __builtin_prefetch(ap + (long)16 * HHk, 0, 0);  // next row-tile
      v8f a0 = {}, a1 = {};
#pragma unroll
      for (int s = 0; s < 16; ++s) {
        v16bf af = cvt_fragA(ap + s * 32);
        a0 = wmma_bf16(af, ldsB(&h2_s[lr][s * 32 + hi * 16]), a0);
        a1 = wmma_bf16(af, ldsB(&h2_s[16 + lr][s * 32 + hi * 16]), a1);
      }
      // Consume tile: th += (S1 + b3_row) * x[bb][d]
      F8 bu, x0u, x1u;
      const float* bp = b3 + r0 + hi * 8;
      bu.q[0] = *(const float4*)bp;       bu.q[1] = *(const float4*)(bp + 4);
      const int d0 = tt * 16 + hi * 8;
      x0u.q[0] = *(const float4*)&x_s[lr][d0];
      x0u.q[1] = *(const float4*)&x_s[lr][d0 + 4];
      x1u.q[0] = *(const float4*)&x_s[16 + lr][d0];
      x1u.q[1] = *(const float4*)&x_s[16 + lr][d0 + 4];
#pragma unroll
      for (int j = 0; j < 8; ++j) {
        th0 = fmaf(a0[j] + bu.f[j], x0u.f[j], th0);
        th1 = fmaf(a1[j] + bu.f[j], x1u.f[j], th1);
      }
    }
    // Combine the two K-half lane groups (rows j vs j+8 covered d%16 halves).
    th0 += __shfl_xor(th0, 16, 32);
    th1 += __shfl_xor(th1, 16, 32);
    if (hi == (w & 1)) {  // owner half of the S2 tile holds row hl in element hl
      float s2v0 = 0.0f, s2v1 = 0.0f;
#pragma unroll
      for (int j = 0; j < 8; ++j)
        if (j == hl) { s2v0 = s2a[j]; s2v1 = s2b[j]; }
      const float fb = b3[tP + I1 + h];
      th_s[h][lr]      = fmaxf(th0 + s2v0 + fb, 0.0f);
      th_s[h][16 + lr] = fmaxf(th1 + s2v1 + fb, 0.0f);
    }
  }

  // ---- fc2 bias tile (S4): rows tP+I3 .. +15, wave 0 publishes to LDS.
  if (w == 0) {
    v8f sa = {}, sb = {};
    const long rb = (tP + I3 + lr) * (long)HHk + sel;
#pragma unroll
    for (int s = 0; s < 16; ++s) {
      v16bf af = cvt_fragA(w3 + rb + s * 32);
      sa = wmma_bf16(af, ldsB(&h2_s[lr][s * 32 + hi * 16]), sa);
      sb = wmma_bf16(af, ldsB(&h2_s[16 + lr][s * 32 + hi * 16]), sb);
    }
#pragma unroll
    for (int j = 0; j < 8; ++j) {
      s4_s[(hi << 3) + j][lr]      = sa[j];
      s4_s[(hi << 3) + j][16 + lr] = sb[j];
    }
  }
  __syncthreads();

  // ---- Phase B: out[b][c] += mean_t( sum_h (fc2_w + b3)*th + fc2_b ). Wave w owns c=2w,2w+1.
  for (int ci = 0; ci < 2; ++ci) {
    const int c = (w << 1) + ci;
    float o0 = 0.0f, o1 = 0.0f;
    for (int tt = 0; tt < 4; ++tt) {
      const long r0 = tP + I2 + (long)c * 64 + tt * 16;
      const float* ap = w3 + (r0 + lr) * HHk + sel;
      v8f a0 = {}, a1 = {};
#pragma unroll
      for (int s = 0; s < 16; ++s) {
        v16bf af = cvt_fragA(ap + s * 32);
        a0 = wmma_bf16(af, ldsB(&h2_s[lr][s * 32 + hi * 16]), a0);
        a1 = wmma_bf16(af, ldsB(&h2_s[16 + lr][s * 32 + hi * 16]), a1);
      }
      F8 bu;
      const float* bp = b3 + r0 + hi * 8;
      bu.q[0] = *(const float4*)bp; bu.q[1] = *(const float4*)(bp + 4);
      const int hh0 = tt * 16 + hi * 8;
#pragma unroll
      for (int j = 0; j < 8; ++j) {
        o0 = fmaf(a0[j] + bu.f[j], th_s[hh0 + j][lr], o0);
        o1 = fmaf(a1[j] + bu.f[j], th_s[hh0 + j][16 + lr], o1);
      }
    }
    o0 += __shfl_xor(o0, 16, 32);
    o1 += __shfl_xor(o1, 16, 32);
    if (hi == 0) {
      const float fb  = b3[tP + I3 + c];
      const float s40 = s4_s[c][lr], s41 = s4_s[c][16 + lr];
      atomicAdd(&out[(long)(b0 + lr) * Cc + c],      (o0 + s40 + fb) * 0.1f);
      atomicAdd(&out[(long)(b0 + 16 + lr) * Cc + c], (o1 + s41 + fb) * 0.1f);
    }
  }
}

extern "C" void kernel_launch(void* const* d_in, const int* in_sizes, int n_in,
                              void* d_out, int out_size, void* d_ws, size_t ws_size,
                              hipStream_t stream) {
  const float* x  = (const float*)d_in[0];
  const float* w1 = (const float*)d_in[1];
  const float* b1 = (const float*)d_in[2];
  const float* w2 = (const float*)d_in[3];
  const float* b2 = (const float*)d_in[4];
  const float* w3 = (const float*)d_in[5];
  const float* b3 = (const float*)d_in[6];
  float* out = (float*)d_out;

  float*  h1 = (float*)d_ws;                                        // 512x512 f32 (1 MB)
  __bf16* h2 = (__bf16*)((char*)d_ws + (size_t)Bb * HHk * 4);       // 512x512 bf16

  zero_f32<<<(out_size + 255) / 256, 256, 0, stream>>>(out, out_size);
  gemm_bias_relu<false><<<128, 256, 0, stream>>>(x,  w1, b1, Dd,  HHk, h1, nullptr);
  gemm_bias_relu<true><<<128, 256, 0, stream>>>(h1, w2, b2, HHk, HHk, nullptr, h2);
  hyper_main<<<(Bb / 32) * Tt, 256, 0, stream>>>(x, w3, b3, h2, out);
}
